// SparseMoE_43508018709041
// MI455X (gfx1250) — matrix operand order repaired and checked by reading, not compiled, not run on hardware
//
#include <hip/hip_runtime.h>
#include <cstdint>
#include <cmath>

// Problem constants (match reference)
#define E_ 8
#define K_ 2
#define D_ 1024
#define H_ 2048
#define N_ 8192

typedef __attribute__((ext_vector_type(16))) __bf16 v16bf;
typedef __attribute__((ext_vector_type(8)))  __bf16 v8bf;
typedef __attribute__((ext_vector_type(8)))  float  v8f;

static __device__ __forceinline__ __bf16 f2bf(float f) { return (__bf16)f; }
static __device__ __forceinline__ float  bf2f(__bf16 b) { return (float)b; }

// ---------------------------------------------------------------------------
// 0. zero the expert counters
// ---------------------------------------------------------------------------
__global__ void init_kernel(int* counts) {
    if (threadIdx.x < E_) counts[threadIdx.x] = 0;
}

// ---------------------------------------------------------------------------
// 1. fp32 -> bf16 elementwise (for x)
// ---------------------------------------------------------------------------
__global__ void cvt_kernel(const float* __restrict__ src, __bf16* __restrict__ dst,
                           unsigned n) {
    unsigned i = blockIdx.x * blockDim.x + threadIdx.x;
    unsigned stride = gridDim.x * blockDim.x;
    for (; i < n; i += stride) dst[i] = f2bf(src[i]);
}

// W1 [E][D][H] fp32  ->  w1t [E][H][D] bf16  (B-operand of GEMM1, K-contiguous)
// i = ((e*H + h)*D + d);  all dims are powers of two -> shift/mask indexing
__global__ void cvt_w1t_kernel(const float* __restrict__ W1, __bf16* __restrict__ w1t) {
    const unsigned n = E_ * D_ * H_;            // 16.8M, fits u32
    unsigned i = blockIdx.x * blockDim.x + threadIdx.x;
    unsigned stride = gridDim.x * blockDim.x;
    for (; i < n; i += stride) {
        unsigned e = i >> 21;                   // / (H*D) = 2^21
        unsigned h = (i >> 10) & (H_ - 1);      // / D % H
        unsigned d = i & (D_ - 1);
        w1t[i] = f2bf(W1[((e << 10) + d) * H_ + h]);
    }
}

// W2 [E][H][D] fp32  ->  w2t [E][D][H] bf16  (B-operand of GEMM2, K-contiguous)
// i = ((e*D + d)*H + h)
__global__ void cvt_w2t_kernel(const float* __restrict__ W2, __bf16* __restrict__ w2t) {
    const unsigned n = E_ * D_ * H_;
    unsigned i = blockIdx.x * blockDim.x + threadIdx.x;
    unsigned stride = gridDim.x * blockDim.x;
    for (; i < n; i += stride) {
        unsigned e = i >> 21;                   // / (D*H)
        unsigned d = (i >> 11) & (D_ - 1);      // / H % D
        unsigned h = i & (H_ - 1);
        w2t[i] = f2bf(W2[((e << 11) + h) * D_ + d]);
    }
}

// ---------------------------------------------------------------------------
// 2. gating: logits = x @ gate_w + gate_b ; top-2 ; softmax ; build expert lists
//    one block per token, 128 threads
// ---------------------------------------------------------------------------
__global__ __launch_bounds__(128) void gate_kernel(
    const float* __restrict__ x, const float* __restrict__ gw,
    const float* __restrict__ gb,
    int* __restrict__ counts, int* __restrict__ tok_list,
    int* __restrict__ tok_e, int* __restrict__ tok_pos, float* __restrict__ tok_w) {
    __shared__ float red[128][E_];
    __shared__ float lg[E_];
    const int n = blockIdx.x;
    const int tid = threadIdx.x;
    float p[E_];
#pragma unroll
    for (int e = 0; e < E_; ++e) p[e] = 0.f;
    const float* xr = x + (long long)n * D_;
    for (int d = tid; d < D_; d += 128) {
        float xv = xr[d];
        const float* g = gw + (long long)d * E_;
#pragma unroll
        for (int e = 0; e < E_; ++e) p[e] += xv * g[e];
    }
#pragma unroll
    for (int e = 0; e < E_; ++e) red[tid][e] = p[e];
    __syncthreads();
    if (tid < E_) {
        float s = gb[tid];
        for (int i = 0; i < 128; ++i) s += red[i][tid];
        lg[tid] = s;
    }
    __syncthreads();
    if (tid == 0) {
        // top-2, lowest index wins ties (matches jax.lax.top_k)
        int i0 = 0; float v0 = lg[0];
#pragma unroll
        for (int e = 1; e < E_; ++e) if (lg[e] > v0) { v0 = lg[e]; i0 = e; }
        int i1 = -1; float v1 = -3.4e38f;
#pragma unroll
        for (int e = 0; e < E_; ++e) if (e != i0 && lg[e] > v1) { v1 = lg[e]; i1 = e; }
        // softmax over {v0, v1}, v0 is the max
        float e1 = __expf(v1 - v0);
        float inv = 1.0f / (1.0f + e1);
        float w0 = inv, w1 = e1 * inv;
        int pos0 = atomicAdd(&counts[i0], 1);
        tok_list[i0 * N_ + pos0] = n;
        tok_e[n * K_ + 0] = i0; tok_pos[n * K_ + 0] = pos0; tok_w[n * K_ + 0] = w0;
        int pos1 = atomicAdd(&counts[i1], 1);
        tok_list[i1 * N_ + pos1] = n;
        tok_e[n * K_ + 1] = i1; tok_pos[n * K_ + 1] = pos1; tok_w[n * K_ + 1] = w1;
    }
}

// ---------------------------------------------------------------------------
// 3. exclusive scan of the 8 counters (single thread; trivial)
// ---------------------------------------------------------------------------
__global__ void scan_kernel(const int* __restrict__ counts, int* __restrict__ offsets) {
    if (threadIdx.x == 0 && blockIdx.x == 0) {
        int acc = 0;
        for (int e = 0; e < E_; ++e) { offsets[e] = acc; acc += counts[e]; }
        offsets[E_] = acc;
    }
}

// ---------------------------------------------------------------------------
// 4. fused expert FFN:  y = relu(xg @ W1[e] + b1[e]) @ W2[e] + b2[e]
//    One block per (expert, 16-token tile). 256 threads = 8 waves.
//    - token rows staged into LDS with GLOBAL_LOAD_ASYNC_TO_LDS_B128 (ASYNCcnt)
//    - GEMMs via v_wmma_f32_16x16x32_bf16 (f32 accumulate)
//    - hidden kept in LDS as bf16, processed in two H/2 passes:
//      LDS = 32KB(x) + 32KB(h) = 64KB
// ---------------------------------------------------------------------------
__global__ __launch_bounds__(256) void ffn_kernel(
    const __bf16* __restrict__ xb,     // [N][D]
    const __bf16* __restrict__ w1t,    // [E][H][D]
    const __bf16* __restrict__ w2t,    // [E][D][H]
    const float*  __restrict__ b1,     // [E][H]
    const float*  __restrict__ b2,     // [E][D]
    const int*    __restrict__ counts, // [E]
    const int*    __restrict__ offsets,// [E+1]
    const int*    __restrict__ tok_list, // [E][N]
    __bf16*       __restrict__ yrows)  // [N*K][D]
{
    __shared__ __bf16 xs[16][D_];      // 32 KB: gathered token tile (A of GEMM1)
    __shared__ __bf16 hs[16][H_ / 2];  // 32 KB: hidden half        (A of GEMM2)

    const int TILES = N_ / 16;         // max tiles per expert
    const int e    = blockIdx.x / TILES;
    const int tile = blockIdx.x % TILES;
    const int cnt  = counts[e];
    if (tile * 16 >= cnt) return;      // block-uniform early exit

    const int tid  = threadIdx.x;
    const int lane = tid & 31;
    const int wave = tid >> 5;
    const int col  = lane & 15;        // N-index inside a 16-wide tile
    const int hiHalf = (lane >= 16);

    // ---- gather 16 token rows of x (bf16) into LDS via async DMA ----------
    // Each thread issues 8 GLOBAL_LOAD_ASYNC_TO_LDS_B128 (16B each); the data
    // never touches VGPRs. Completion tracked with ASYNCcnt.
    {
        const unsigned lds_base = (unsigned)(uintptr_t)&xs[0][0];
        for (int i = tid; i < 16 * (D_ / 8); i += 256) {
            int r  = i >> 7;           // row 0..15  (D_/8 = 128 chunks per row)
            int c8 = i & 127;
            int p  = tile * 16 + r; if (p >= cnt) p = cnt - 1;
            int tok = tok_list[e * N_ + p];
            const __bf16* src = xb + (long long)tok * D_ + c8 * 8;
            unsigned dst = lds_base + (unsigned)i * 16u;  // contiguous 16B chunks
            asm volatile("global_load_async_to_lds_b128 %0, %1, off"
                         :: "v"(dst), "v"(src) : "memory");
        }
        asm volatile("s_wait_asynccnt 0" ::: "memory");
    }
    __syncthreads();

    // persistent f32 accumulators for the 8 output tiles this wave owns
    v8f acc[8];
#pragma unroll
    for (int t = 0; t < 8; ++t) acc[t] = (v8f){0,0,0,0,0,0,0,0};

    const int abase_off = hiHalf ? 16 : 0;  // A-fragment K offset, contiguous order k = 16*half + s
    const int bbase_off = hiHalf ? 16 : 0;  // B-fragment K offset, contiguous order k = 16*half + s

    for (int hb = 0; hb < 2; ++hb) {
        // ---------------- GEMM1: h(:, half) = relu(xs @ W1 + b1) -----------
        for (int t = 0; t < 8; ++t) {
            const int jh   = wave + 8 * t;              // 64 col-tiles / half
            const int hcol = hb * (H_ / 2) + jh * 16 + col;
            const __bf16* bp = w1t + ((long long)e * H_ + hcol) * D_;
            v8f c = (v8f){0,0,0,0,0,0,0,0};
#pragma unroll 4
            for (int kb = 0; kb < D_; kb += 32) {
                // A fragment (16x32 bf16) from LDS
                v8bf lo = *(const v8bf*)&xs[col][kb + abase_off];
                v8bf hi = *(const v8bf*)&xs[col][kb + abase_off + 8];
                v16bf a = __builtin_shufflevector(lo, hi, 0,1,2,3,4,5,6,7,
                                                          8,9,10,11,12,13,14,15);
                // B fragment (32x16 bf16): 16 contiguous K elems per lane
                v16bf b = *(const v16bf*)(bp + kb + bbase_off);
                __builtin_prefetch(bp + kb + 256, 0, 0);
                c = __builtin_amdgcn_wmma_f32_16x16x32_bf16(
                        false, a, false, b, (short)0, c, false, false);
            }
            const float bias = b1[e * H_ + hcol];
#pragma unroll
            for (int i = 0; i < 8; ++i) {
                int r = i + (hiHalf ? 8 : 0);
                float v = c[i] + bias;
                hs[r][jh * 16 + col] = f2bf(v > 0.f ? v : 0.f);
            }
        }
        __syncthreads();
        // ---------------- GEMM2 partial: acc += h_half @ W2(half,:) --------
        for (int t = 0; t < 8; ++t) {
            const int jo   = wave + 8 * t;              // 64 col-tiles of D
            const int dcol = jo * 16 + col;
            const __bf16* bp = w2t + ((long long)e * D_ + dcol) * H_ + hb * (H_ / 2);
            v8f c = acc[t];
#pragma unroll 4
            for (int kb = 0; kb < H_ / 2; kb += 32) {
                v8bf lo = *(const v8bf*)&hs[col][kb + abase_off];
                v8bf hi = *(const v8bf*)&hs[col][kb + abase_off + 8];
                v16bf a = __builtin_shufflevector(lo, hi, 0,1,2,3,4,5,6,7,
                                                          8,9,10,11,12,13,14,15);
                v16bf b = *(const v16bf*)(bp + kb + bbase_off);
                __builtin_prefetch(bp + kb + 256, 0, 0);
                c = __builtin_amdgcn_wmma_f32_16x16x32_bf16(
                        false, a, false, b, (short)0, c, false, false);
            }
            acc[t] = c;
        }
        __syncthreads();
    }

    // ---- epilogue: + b2, store this tile's rows into the per-slot buffer ---
    const int base = offsets[e];
#pragma unroll
    for (int t = 0; t < 8; ++t) {
        const int jo   = wave + 8 * t;
        const int dcol = jo * 16 + col;
        const float bias = b2[e * D_ + dcol];
#pragma unroll
        for (int i = 0; i < 8; ++i) {
            int r = i + (hiHalf ? 8 : 0);
            int p = tile * 16 + r;
            if (p < cnt)
                yrows[(long long)(base + p) * D_ + dcol] = f2bf(acc[t][i] + bias);
        }
    }
}

// ---------------------------------------------------------------------------
// 5. deterministic weighted combine: out[n] = w0*y[slot0] + w1*y[slot1]
// ---------------------------------------------------------------------------
__global__ __launch_bounds__(256) void combine_kernel(
    const __bf16* __restrict__ yrows, const int* __restrict__ offsets,
    const int* __restrict__ tok_e, const int* __restrict__ tok_pos,
    const float* __restrict__ tok_w, float* __restrict__ out) {
    const int n = blockIdx.x;
    const long long s0 = (long long)(offsets[tok_e[n*K_+0]] + tok_pos[n*K_+0]) * D_;
    const long long s1 = (long long)(offsets[tok_e[n*K_+1]] + tok_pos[n*K_+1]) * D_;
    const float w0 = tok_w[n*K_+0], w1 = tok_w[n*K_+1];
    for (int d = threadIdx.x; d < D_; d += 256)
        out[(long long)n * D_ + d] = w0 * bf2f(yrows[s0 + d]) + w1 * bf2f(yrows[s1 + d]);
}

// ---------------------------------------------------------------------------
// launch
// ---------------------------------------------------------------------------
extern "C" void kernel_launch(void* const* d_in, const int* in_sizes, int n_in,
                              void* d_out, int out_size, void* d_ws, size_t ws_size,
                              hipStream_t stream) {
    const float* x   = (const float*)d_in[0];
    const float* gw  = (const float*)d_in[1];
    const float* gb  = (const float*)d_in[2];
    const float* W1  = (const float*)d_in[3];
    const float* b1  = (const float*)d_in[4];
    const float* W2  = (const float*)d_in[5];
    const float* b2  = (const float*)d_in[6];
    float* out = (float*)d_out;

    // --- carve workspace (total ~118 MB) ---
    uintptr_t p = (uintptr_t)d_ws;
    auto carve = [&](size_t bytes) -> void* {
        void* r = (void*)p;
        p += (bytes + 255) & ~(size_t)255;
        return r;
    };
    __bf16* xb   = (__bf16*)carve((size_t)N_ * D_ * 2);          // 16.8 MB
    __bf16* w1t  = (__bf16*)carve((size_t)E_ * H_ * D_ * 2);     // 33.6 MB
    __bf16* w2t  = (__bf16*)carve((size_t)E_ * D_ * H_ * 2);     // 33.6 MB
    __bf16* yrows= (__bf16*)carve((size_t)N_ * K_ * D_ * 2);     // 33.6 MB
    int* counts  = (int*)carve(E_ * sizeof(int));
    int* offsets = (int*)carve((E_ + 1) * sizeof(int));
    int* tok_list= (int*)carve((size_t)E_ * N_ * sizeof(int));   // 256 KB
    int* tok_e   = (int*)carve((size_t)N_ * K_ * sizeof(int));
    int* tok_pos = (int*)carve((size_t)N_ * K_ * sizeof(int));
    float* tok_w = (float*)carve((size_t)N_ * K_ * sizeof(float));

    init_kernel<<<1, 32, 0, stream>>>(counts);

    cvt_kernel<<<2048, 256, 0, stream>>>(x, xb, (unsigned)(N_ * D_));
    cvt_w1t_kernel<<<4096, 256, 0, stream>>>(W1, w1t);
    cvt_w2t_kernel<<<4096, 256, 0, stream>>>(W2, w2t);

    gate_kernel<<<N_, 128, 0, stream>>>(x, gw, gb, counts, tok_list,
                                        tok_e, tok_pos, tok_w);
    scan_kernel<<<1, 32, 0, stream>>>(counts, offsets);

    ffn_kernel<<<E_ * (N_ / 16), 256, 0, stream>>>(
        xb, w1t, w2t, b1, b2, counts, offsets, tok_list, yrows);

    combine_kernel<<<N_, 256, 0, stream>>>(yrows, offsets, tok_e, tok_pos,
                                           tok_w, out);
}
